// SegMambaBackbone2D_58265526337689
// MI455X (gfx1250) — matrix operand order repaired and checked by reading, not compiled, not run on hardware
//
#include <hip/hip_runtime.h>

// ---------------------------------------------------------------------------
// Types for CDNA5 WMMA (gfx1250, wave32)
// ---------------------------------------------------------------------------
typedef unsigned short u16;
typedef __attribute__((ext_vector_type(16))) __bf16        v16bf;
typedef __attribute__((ext_vector_type(8)))  float         v8f;
typedef __attribute__((ext_vector_type(4)))  unsigned int  u32x4;

union FragU { v16bf v; u32x4 q[2]; };

__device__ __forceinline__ u16 f2bf(float f) {
  union { float f; unsigned int u; } c; c.f = f;
  unsigned int u = c.u;
  unsigned int r = u + 0x7FFFu + ((u >> 16) & 1u); // round-to-nearest-even
  return (u16)(r >> 16);
}

#define CDIV(a, b) (((a) + (b) - 1) / (b))

// ---------------------------------------------------------------------------
// Generic strided GEMM with bf16 WMMA, f32 accumulate.
//   C[m,n] (+)= act( sum_k A[m,k] * B[k,n] + bias[m] )
// A is row-major [M x K] (weights). B addressed as B[k*sbk + n*sbn].
// C addressed as C[m*scm + n*scn]. Per-batch strides batB/batC (batA == 0).
// act: 0 none, 1 softplus, 2 exact GELU.  addC: accumulate into existing C.
// Block = 128 threads (4 waves). Tile 64(M) x 128(N), K-step 32.
// Each wave owns a 32x64 sub-tile = 2x4 WMMA accumulators (8 wmma / K-step).
// ---------------------------------------------------------------------------
__global__ __launch_bounds__(128) void k_gemm_wmma(
    const float* __restrict__ A, long lda,
    const float* __restrict__ B, long sbk, long sbn, long batB,
    float* __restrict__ C, long scm, long scn, long batC,
    int M, int N, int K, const float* __restrict__ bias, int act, int addC) {
  __shared__ alignas(16) u16 As[64][40];    // [m][k]
  __shared__ alignas(16) u16 Bs[128][40];   // [n][k]  (B transposed in LDS)
  const int bz = blockIdx.z;
  B += (long)bz * batB;
  C += (long)bz * batC;
  const int m0 = blockIdx.y * 64, n0 = blockIdx.x * 128;
  const int tid = threadIdx.x, lane = tid & 31, wave = tid >> 5;
  const int wm = (wave >> 1) * 32, wn = (wave & 1) * 64;

  v8f zero = {0.f, 0.f, 0.f, 0.f, 0.f, 0.f, 0.f, 0.f};
  v8f acc[2][4];
  for (int i = 0; i < 2; i++)
    for (int j = 0; j < 4; j++) acc[i][j] = zero;

  for (int kk = 0; kk < K; kk += 32) {
    // A tile: 64x32 elements, f32 -> bf16
    for (int i = tid; i < 64 * 32; i += 128) {
      int r = i >> 5, c = i & 31;
      int gm = m0 + r, gk = kk + c;
      float va = (gm < M && gk < K) ? A[(long)gm * lda + gk] : 0.f;
      As[r][c] = f2bf(va);
    }
    // B tile: 128x32 elements (stored [n][k])
    for (int i = tid; i < 128 * 32; i += 128) {
      int r = i >> 5, c = i & 31;
      int gn = n0 + r, gk = kk + c;
      float vb = (gn < N && gk < K) ? B[(long)gk * sbk + (long)gn * sbn] : 0.f;
      Bs[r][c] = f2bf(vb);
    }
    __syncthreads();

    const int rlo = lane & 15;
    const int hib = (lane >> 4) * 8;    // A: K-offset base per half-wave
    const int koff = (lane >> 4) * 8;   // B: K-offset base per half-wave
    FragU fa[2], fb[4];
    for (int i = 0; i < 2; i++) {
      int row = wm + 16 * i + rlo;
      fa[i].q[0] = *(const u32x4*)&As[row][hib];
      fa[i].q[1] = *(const u32x4*)&As[row][hib + 16];
    }
    for (int j = 0; j < 4; j++) {
      int col = wn + 16 * j + rlo;
      fb[j].q[0] = *(const u32x4*)&Bs[col][koff];
      fb[j].q[1] = *(const u32x4*)&Bs[col][koff + 16];
    }
    for (int i = 0; i < 2; i++)
      for (int j = 0; j < 4; j++)
        acc[i][j] = __builtin_amdgcn_wmma_f32_16x16x32_bf16(
            false, fa[i].v, false, fb[j].v, (short)0, acc[i][j], false, false);
    __syncthreads();
  }

  const int nloc = lane & 15, mhi = (lane >> 4) * 8;
  for (int i = 0; i < 2; i++)
    for (int j = 0; j < 4; j++) {
      int n = n0 + wn + 16 * j + nloc;
      if (n >= N) continue;
      for (int r = 0; r < 8; r++) {
        int m = m0 + wm + 16 * i + mhi + r;
        if (m >= M) continue;
        float v = acc[i][j][r];
        if (bias) v += bias[m];
        if (act == 1) {  // softplus (stable)
          float ax = fabsf(v);
          v = log1pf(expf(-ax)) + fmaxf(v, 0.f);
        } else if (act == 2) {  // exact GELU
          v = 0.5f * v * (1.f + erff(v * 0.70710678118654752f));
        }
        long o = (long)m * scm + (long)n * scn;
        if (addC) v += C[o];
        C[o] = v;
      }
    }
}

// ---------------------------------------------------------------------------
// Direct conv2d NCHW, one thread per output element.
// ---------------------------------------------------------------------------
__global__ void k_conv2d(const float* __restrict__ in, const float* __restrict__ w,
                         const float* __restrict__ bias, float* __restrict__ out,
                         int Bn, int Ci, int Hi, int Wi, int Co, int Ho, int Wo,
                         int KH, int KW, int stride, int pad) {
  long i = (long)blockIdx.x * blockDim.x + threadIdx.x;
  long total = (long)Bn * Co * Ho * Wo;
  if (i >= total) return;
  int wo = (int)(i % Wo); long t = i / Wo;
  int ho = (int)(t % Ho); t /= Ho;
  int co = (int)(t % Co); int b = (int)(t / Co);
  float s = bias[co];
  for (int ci = 0; ci < Ci; ci++) {
    const float* ip = in + (((long)b * Ci + ci) * Hi) * Wi;
    const float* wp = w + (((long)co * Ci + ci) * KH) * KW;
    for (int kh = 0; kh < KH; kh++) {
      int hi = ho * stride - pad + kh;
      if (hi < 0 || hi >= Hi) continue;
      for (int kw = 0; kw < KW; kw++) {
        int wi = wo * stride - pad + kw;
        if (wi < 0 || wi >= Wi) continue;
        s += wp[kh * KW + kw] * ip[(long)hi * Wi + wi];
      }
    }
  }
  out[i] = s;
}

// ---------------------------------------------------------------------------
// Instance norm over one (b,c) row of HW elements. Block=256 per row.
// relu: apply max(0,.) after norm. Supports in == out.
// ---------------------------------------------------------------------------
__global__ __launch_bounds__(256) void k_instnorm(const float* __restrict__ in,
                                                  float* __restrict__ out, int HW,
                                                  int relu) {
  const float* p = in + (long)blockIdx.x * HW;
  float* q = out + (long)blockIdx.x * HW;
  __shared__ float s1[256], s2[256];
  float a = 0.f, b = 0.f;
  for (int i = threadIdx.x; i < HW; i += 256) { float v = p[i]; a += v; b += v * v; }
  s1[threadIdx.x] = a; s2[threadIdx.x] = b;
  __syncthreads();
  for (int off = 128; off > 0; off >>= 1) {
    if ((int)threadIdx.x < off) { s1[threadIdx.x] += s1[threadIdx.x + off]; s2[threadIdx.x] += s2[threadIdx.x + off]; }
    __syncthreads();
  }
  float mu = s1[0] / HW;
  float var = s2[0] / HW - mu * mu;
  float inv = rsqrtf(var + 1e-5f);
  for (int i = threadIdx.x; i < HW; i += 256) {
    float v = (p[i] - mu) * inv;
    if (relu) v = fmaxf(v, 0.f);
    q[i] = v;
  }
}

// layer_norm over channel dim with transpose: x[B,C,L] -> t[B,L,C]
__global__ void k_lnt(const float* __restrict__ x, const float* __restrict__ w,
                      const float* __restrict__ bsh, float* __restrict__ t,
                      int Bn, int C, int L) {
  long i = (long)blockIdx.x * blockDim.x + threadIdx.x;
  if (i >= (long)Bn * L) return;
  int b = (int)(i / L), l = (int)(i % L);
  const float* xp = x + (long)b * C * L + l;
  float mu = 0.f;
  for (int c = 0; c < C; c++) mu += xp[(long)c * L];
  mu /= C;
  float var = 0.f;
  for (int c = 0; c < C; c++) { float d = xp[(long)c * L] - mu; var += d * d; }
  var /= C;
  float inv = rsqrtf(var + 1e-5f);
  float* tp = t + ((long)b * L + l) * C;
  for (int c = 0; c < C; c++) tp[c] = (xp[(long)c * L] - mu) * inv * w[c] + bsh[c];
}

// depthwise causal conv1d (k=4, left pad 3) + SiLU over first di channels of xz[B,E,L]
__global__ void k_cconv_silu(const float* __restrict__ xz, const float* __restrict__ w,
                             const float* __restrict__ b, float* __restrict__ xc,
                             int Bn, int E, int di, int L) {
  long i = (long)blockIdx.x * blockDim.x + threadIdx.x;
  if (i >= (long)Bn * di * L) return;
  int l = (int)(i % L); long t = i / L;
  int d = (int)(t % di); int bb = (int)(t / di);
  const float* xp = xz + ((long)bb * E + d) * L;
  float s = b[d];
  for (int j = 0; j < 4; j++) {
    int ll = l - 3 + j;
    if (ll >= 0) s += w[d * 4 + j] * xp[ll];
  }
  s = s / (1.f + expf(-s));  // silu
  xc[((long)bb * di + d) * L + l] = s;
}

// reverse along L:  out[row,l] = in[row, L-1-l]
__global__ void k_rev(float* __restrict__ out, const float* __restrict__ in,
                      long rows, int L) {
  long i = (long)blockIdx.x * blockDim.x + threadIdx.x;
  if (i >= rows * L) return;
  long row = i / L; int l = (int)(i % L);
  out[i] = in[row * L + (L - 1 - l)];
}

// slice permute: out[row, j*ns+s] = in[row, s*(L/ns)+j]
__global__ void k_slc(float* __restrict__ out, const float* __restrict__ in,
                      long rows, int L, int ns) {
  long i = (long)blockIdx.x * blockDim.x + threadIdx.x;
  if (i >= rows * L) return;
  long row = i / L; int lp = (int)(i % L);
  int Lns = L / ns;
  int s = lp % ns, j = lp / ns;
  out[i] = in[row * L + (long)s * Lns + j];
}

// ---------------------------------------------------------------------------
// Selective scan. One 16-lane half-wave per (b,d): lane = state n.
// h_n <- exp(dt*A[d,n])*h_n + dt*B[n,l]*u ; y = sum_n h_n*C[n,l] + D*u ; *silu(z)
// Cross-lane sum via 4 xor-shuffles (stays inside each 16-lane group, wave32).
// ---------------------------------------------------------------------------
__global__ __launch_bounds__(128) void k_scan(
    const float* __restrict__ xc, const float* __restrict__ dt,
    const float* __restrict__ xdbl, const float* __restrict__ A_log,
    const float* __restrict__ Dp, const float* __restrict__ xzb,
    float* __restrict__ out, int Bn, int di, int E, int R, int L) {
  int gt = blockIdx.x * blockDim.x + threadIdx.x;
  int wave = gt >> 5, lane = gt & 31;
  int idx = wave * 2 + (lane >> 4);
  if (idx >= Bn * di) return;
  int b = idx / di, d = idx % di;
  int n = lane & 15;
  float Ac = -expf(A_log[d * 16 + n]);
  const float* dtp = dt + ((long)b * di + d) * L;
  const float* up = xc + ((long)b * di + d) * L;
  const float* bp = xdbl + ((long)b * (R + 32) + R + n) * L;
  const float* cp = xdbl + ((long)b * (R + 32) + R + 16 + n) * L;
  const float* zp = xzb + ((long)b * E + di + d) * L;
  float* op = out + ((long)b * di + d) * L;
  float Dd = Dp[d];
  float h = 0.f;
  for (int l = 0; l < L; l++) {
    float delta = dtp[l], u = up[l];
    h = expf(delta * Ac) * h + delta * bp[l] * u;
    float part = h * cp[l];
    part += __shfl_xor(part, 1);
    part += __shfl_xor(part, 2);
    part += __shfl_xor(part, 4);
    part += __shfl_xor(part, 8);
    if (n == 0) {
      float z = zp[l];
      float y = part + Dd * u;
      op[l] = y * (z / (1.f + expf(-z)));
    }
  }
}

// accumulate branch output into ysum with L-remapping.
// mode 0: store identity. mode 1: += reversed. mode 2: += slice-inverse (ns).
__global__ void k_acc(float* __restrict__ ysum, const float* __restrict__ ytmp,
                      int Bn, int di, int L, int mode, int ns) {
  long i = (long)blockIdx.x * blockDim.x + threadIdx.x;
  if (i >= (long)Bn * di * L) return;
  int l = (int)(i % L);
  long base = (i / L) * (long)L;
  if (mode == 0) {
    ysum[i] = ytmp[i];
  } else if (mode == 1) {
    ysum[i] += ytmp[base + (L - 1 - l)];
  } else {
    int Lns = L / ns;
    int s = l / Lns, j = l % Lns;
    ysum[i] += ytmp[base + (long)j * ns + s];
  }
}

__global__ void k_add(float* __restrict__ dst, const float* __restrict__ src, long n) {
  long i = (long)blockIdx.x * blockDim.x + threadIdx.x;
  if (i < n) dst[i] += src[i];
}
__global__ void k_copy(float* __restrict__ dst, const float* __restrict__ src, long n) {
  long i = (long)blockIdx.x * blockDim.x + threadIdx.x;
  if (i < n) dst[i] = src[i];
}

// ---------------------------------------------------------------------------
// Host orchestration
// ---------------------------------------------------------------------------
struct BranchP { const float *A_log, *D, *conv_b, *conv_w, *dtproj_b, *dtproj_w, *xproj_w; };
struct LayerP  { BranchP bwd, fwd, slc; const float *in_proj_w, *ln_b, *ln_w, *out_proj_w; };

extern "C" void kernel_launch(void* const* d_in, const int* in_sizes, int n_in,
                              void* d_out, int out_size, void* d_ws, size_t ws_size,
                              hipStream_t stream) {
  (void)in_sizes; (void)n_in; (void)out_size; (void)ws_size;
  const int Bn = 2;
  const int DIMS_[4] = {48, 96, 192, 384};
  const int NS_[4] = {32, 16, 8, 4};

  // ---- unpack inputs in JAX tree-flatten order (sorted dict keys) ----
  int idx = 0;
  auto P = [&]() { return (const float*)d_in[idx++]; };
  const float *down_b[4], *down_w[4];
  for (int i = 0; i < 4; i++) { down_b[i] = P(); down_w[i] = P(); }
  struct GscP { const float *b[4], *w[4]; } g[4];
  for (int i = 0; i < 4; i++)
    for (int j = 0; j < 4; j++) { g[i].b[j] = P(); g[i].w[j] = P(); }  // c1..c4: b, w
  struct MlpP { const float *fc1_b, *fc1_w, *fc2_b, *fc2_w; } mlp[4];
  for (int i = 0; i < 4; i++) { mlp[i].fc1_b = P(); mlp[i].fc1_w = P(); mlp[i].fc2_b = P(); mlp[i].fc2_w = P(); }
  auto parse_branch = [&](BranchP& br) {
    br.A_log = P(); br.D = P(); br.conv_b = P(); br.conv_w = P();
    br.dtproj_b = P(); br.dtproj_w = P(); br.xproj_w = P();
  };
  LayerP layers[4][2];
  for (int s = 0; s < 4; s++)
    for (int ly = 0; ly < 2; ly++) {
      LayerP& lp = layers[s][ly];
      parse_branch(lp.bwd);
      parse_branch(lp.fwd);
      lp.in_proj_w = P(); lp.ln_b = P(); lp.ln_w = P(); lp.out_proj_w = P();
      parse_branch(lp.slc);
    }
  const float* xin = P();

  // ---- workspace (sized for stage 0 maxima; ~87.5 MB) ----
  float* w = (float*)d_ws;
  auto alloc = [&](long n) { float* p = w; w += n; return p; };
  float* xA   = alloc(884736);
  float* xB   = alloc(884736);
  float* tb   = alloc(884736);
  float* xz   = alloc(3538944);
  float* xzp  = alloc(3538944);
  float* xc   = alloc(1769472);
  float* xdbl = alloc(645120);
  float* dtb  = alloc(1769472);
  float* ytmp = alloc(1769472);
  float* ysum = alloc(1769472);
  float* h1   = alloc(1769472);
  float* g1   = alloc(884736);
  float* g2   = alloc(884736);
  float* g3   = alloc(884736);
  float* out  = (float*)d_out;

  auto gemm = [&](const float* A, long lda, const float* B, long sbk, long sbn,
                  long batB, float* Cm, long scm, long scn, long batC, int M, int N,
                  int K, const float* bias, int act, int addC) {
    dim3 grid(CDIV(N, 128), CDIV(M, 64), Bn);
    k_gemm_wmma<<<grid, 128, 0, stream>>>(A, lda, B, sbk, sbn, batB, Cm, scm, scn,
                                          batC, M, N, K, bias, act, addC);
  };

  auto run_branch = [&](const float* xzin, const BranchP& br, int C, int L) {
    int di = 2 * C, E = 4 * C, R = (C + 15) / 16;
    long tot = (long)Bn * di * L;
    k_cconv_silu<<<CDIV(tot, 256), 256, 0, stream>>>(xzin, br.conv_w, br.conv_b, xc, Bn, E, di, L);
    // xdbl[b,r,l] = xproj_w[r,:] . xc[b,:,l]
    gemm(br.xproj_w, di, xc, L, 1, (long)di * L, xdbl, L, 1, (long)(R + 32) * L,
         R + 32, L, di, nullptr, 0, 0);
    // dt[b,d,l] = softplus(dtproj_w[d,:] . xdbl[b,0:R,l] + dtproj_b[d])
    gemm(br.dtproj_w, R, xdbl, L, 1, (long)(R + 32) * L, dtb, L, 1, (long)di * L,
         di, L, R, br.dtproj_b, 1, 0);
    int waves = Bn * di / 2;
    k_scan<<<CDIV(waves * 32, 128), 128, 0, stream>>>(xc, dtb, xdbl, br.A_log,
                                                      br.D, xzin, ytmp, Bn, di, E, R, L);
  };

  auto mamba = [&](float* x, const LayerP& lp, int C, int L, int ns) {
    int di = 2 * C, E = 4 * C;
    k_lnt<<<CDIV((long)Bn * L, 256), 256, 0, stream>>>(x, lp.ln_w, lp.ln_b, tb, Bn, C, L);
    // xz[b,e,l] = in_proj_w[e,:] . t[b,l,:]
    gemm(lp.in_proj_w, C, tb, 1, C, (long)L * C, xz, L, 1, (long)E * L, E, L, C,
         nullptr, 0, 0);
    long tot = (long)Bn * di * L;
    long totE = (long)Bn * E * L;
    // forward branch
    run_branch(xz, lp.fwd, C, L);
    k_acc<<<CDIV(tot, 256), 256, 0, stream>>>(ysum, ytmp, Bn, di, L, 0, ns);
    // backward branch (reversed L)
    k_rev<<<CDIV(totE, 256), 256, 0, stream>>>(xzp, xz, (long)Bn * E, L);
    run_branch(xzp, lp.bwd, C, L);
    k_acc<<<CDIV(tot, 256), 256, 0, stream>>>(ysum, ytmp, Bn, di, L, 1, ns);
    // slice branch (interleaved L)
    k_slc<<<CDIV(totE, 256), 256, 0, stream>>>(xzp, xz, (long)Bn * E, L, ns);
    run_branch(xzp, lp.slc, C, L);
    k_acc<<<CDIV(tot, 256), 256, 0, stream>>>(ysum, ytmp, Bn, di, L, 2, ns);
    // x[b,c,l] += out_proj_w[c,:] . ysum[b,:,l]
    gemm(lp.out_proj_w, di, ysum, L, 1, (long)di * L, x, L, 1, (long)C * L, C, L,
         di, nullptr, 0, 1);
  };

  float* cur = xA;
  float* oth = xB;
  int H = 0, prevC = 0, prevH = 0;
  long outoff = 0;

  for (int s = 0; s < 4; s++) {
    int C = DIMS_[s];
    if (s == 0) {
      H = 96;
      long tot = (long)Bn * 48 * 96 * 96;
      k_conv2d<<<CDIV(tot, 256), 256, 0, stream>>>(xin, down_w[0], down_b[0], cur,
                                                   Bn, 1, 192, 192, 48, 96, 96, 7, 7, 2, 3);
    } else {
      k_instnorm<<<Bn * prevC, 256, 0, stream>>>(cur, cur, prevH * prevH, 0);
      int Ho = prevH / 2;
      long tot = (long)Bn * C * Ho * Ho;
      k_conv2d<<<CDIV(tot, 256), 256, 0, stream>>>(cur, down_w[s], down_b[s], oth,
                                                   Bn, prevC, prevH, prevH, C, Ho, Ho, 2, 2, 2, 0);
      float* t2 = cur; cur = oth; oth = t2;
      H = Ho;
    }
    int L = H * H;
    long nCL = (long)Bn * C * L;

    // ---- GSC block ----
    k_conv2d<<<CDIV(nCL, 256), 256, 0, stream>>>(cur, g[s].w[0], g[s].b[0], g1,
                                                 Bn, C, H, H, C, H, H, 3, 3, 1, 1);
    k_instnorm<<<Bn * C, 256, 0, stream>>>(g1, g1, L, 1);
    k_conv2d<<<CDIV(nCL, 256), 256, 0, stream>>>(g1, g[s].w[1], g[s].b[1], g2,
                                                 Bn, C, H, H, C, H, H, 3, 3, 1, 1);
    k_instnorm<<<Bn * C, 256, 0, stream>>>(g2, g2, L, 1);
    k_conv2d<<<CDIV(nCL, 256), 256, 0, stream>>>(cur, g[s].w[2], g[s].b[2], g3,
                                                 Bn, C, H, H, C, H, H, 1, 1, 1, 0);
    k_instnorm<<<Bn * C, 256, 0, stream>>>(g3, g3, L, 1);
    k_add<<<CDIV(nCL, 256), 256, 0, stream>>>(g2, g3, nCL);      // x1 + x2
    k_conv2d<<<CDIV(nCL, 256), 256, 0, stream>>>(g2, g[s].w[3], g[s].b[3], g1,
                                                 Bn, C, H, H, C, H, H, 1, 1, 1, 0);
    k_instnorm<<<Bn * C, 256, 0, stream>>>(g1, g1, L, 1);
    k_add<<<CDIV(nCL, 256), 256, 0, stream>>>(cur, g1, nCL);     // y + residual

    // ---- mamba layers ----
    for (int ly = 0; ly < 2; ly++) mamba(cur, layers[s][ly], C, L, NS_[s]);

    // ---- stage head: instance_norm + channel MLP, into d_out segment ----
    k_instnorm<<<Bn * C, 256, 0, stream>>>(cur, g1, L, 0);
    k_copy<<<CDIV(nCL, 256), 256, 0, stream>>>(out + outoff, g1, nCL);
    gemm(mlp[s].fc1_w, C, g1, L, 1, (long)C * L, h1, L, 1, (long)2 * C * L,
         2 * C, L, C, mlp[s].fc1_b, 2, 0);                         // GELU
    gemm(mlp[s].fc2_w, 2 * C, h1, L, 1, (long)2 * C * L, out + outoff, L, 1,
         (long)C * L, C, L, 2 * C, mlp[s].fc2_b, 0, 1);            // += residual
    outoff += nCL;
    prevC = C; prevH = H;
  }
}
